// MultiHeadSelfAttentionLayer_39754217292499
// MI455X (gfx1250) — matrix operand order repaired and checked by reading, not compiled, not run on hardware
//
#include <hip/hip_runtime.h>
#include <hip/hip_bf16.h>

// ---------------------------------------------------------------------------
// Multi-head self-attention: QKV proj (WMMA f16) -> flash attention (WMMA f16)
// -> output proj (WMMA f16, fp32 out). MI455X / gfx1250, wave32.
// Round 3: async global->LDS staging (ASYNCcnt) for pure-copy tiles (K tile in
// attention, f16 A tile in final GEMM), double-buffered; DPP softmax kept.
// ---------------------------------------------------------------------------

typedef __attribute__((ext_vector_type(16))) _Float16 v16h;
typedef __attribute__((ext_vector_type(8)))  _Float16 v8h;
typedef __attribute__((ext_vector_type(4)))  _Float16 v4h;
typedef __attribute__((ext_vector_type(8)))  float    v8f;
typedef __attribute__((ext_vector_type(4)))  float    v4f;
typedef __attribute__((ext_vector_type(2)))  float    v2f;

#define EMB   1024          // E == H*D == KV_OUT
#define SEQ   2048
#define NB    2
#define HEADS 16
#define HDIM  64
#define MROWS 4096          // NB*SEQ

static __device__ __forceinline__ v16h cat8(v8h lo, v8h hi) {
    v16h r;
#pragma unroll
    for (int i = 0; i < 8; ++i) { r[i] = lo[i]; r[i + 8] = hi[i]; }
    return r;
}

static __device__ __forceinline__ v8f wmma_f16(v16h a, v16h b, v8f c) {
    return __builtin_amdgcn_wmma_f32_16x16x32_f16(
        false, a, false, b, (short)0, c, false, false);
}

// ---- DPP cross-lane reductions within each 16-lane half (VALU only) -------
template <int CTRL>
static __device__ __forceinline__ float dpp_movf(float x) {
    return __int_as_float(__builtin_amdgcn_update_dpp(
        0, __float_as_int(x), CTRL, 0xF, 0xF, true));
}
static __device__ __forceinline__ float row16_max(float x) {
    x = fmaxf(x, dpp_movf<0xB1>(x));    // quad_perm xor1
    x = fmaxf(x, dpp_movf<0x4E>(x));    // quad_perm xor2
    x = fmaxf(x, dpp_movf<0x164>(x));   // row_xmask:4
    x = fmaxf(x, dpp_movf<0x168>(x));   // row_xmask:8
    return x;
}
static __device__ __forceinline__ float row16_sum(float x) {
    x += dpp_movf<0xB1>(x);
    x += dpp_movf<0x4E>(x);
    x += dpp_movf<0x164>(x);
    x += dpp_movf<0x168>(x);
    return x;
}

// ---------------------------------------------------------------------------
// GEMM: out[4096,1024] = A[4096,1024] @ W[1024,1024] + bias, * scale
// Block = 256 thr (8 waves), tile 128x64, k-step 32.
// A_F16 path: A tile staged with async global->LDS b128 (double-buffered).
// ---------------------------------------------------------------------------
template <bool A_F16, bool OUT_F32>
__global__ __launch_bounds__(256) void gemm_kernel(
    const void* __restrict__ Ag, const float* __restrict__ Wg,
    const float* __restrict__ bias, void* __restrict__ Og, float scale)
{
    constexpr int ASH  = 128 * 40;       // [m][k] f16, pad 40/row
    constexpr int ABUF = A_F16 ? 2 : 1;  // async path double-buffers A
    constexpr int BSH  = 64 * 40;        // W^T [n][k] f16, pad 40/row
    constexpr size_t EPI  = OUT_F32 ? (size_t)8192 * 4 : (size_t)8192 * 2;
    constexpr size_t MAIN = (size_t)(ASH * ABUF + BSH) * 2;
    constexpr size_t POOLB = (MAIN > EPI) ? MAIN : EPI;
    __shared__ __align__(16) char pool[POOLB];
    _Float16* Ash = (_Float16*)pool;
    _Float16* Bsh = Ash + ASH * ABUF;

    const int tid  = threadIdx.x;
    const int wave = tid >> 5;
    const int lane = tid & 31;
    const int lg   = lane >> 4;
    const int ln   = lane & 15;
    const int m0   = blockIdx.x * 128;
    const int n0   = blockIdx.y * 64;

    v8f acc[4] = {};
    v4f pfA4[4];
    v2f pfW[4];

    auto issueA = [&](int k0, int buf) {   // async f16 A tile -> LDS
        const _Float16* abase = (const _Float16*)Ag;
#pragma unroll
        for (int i = 0; i < 2; ++i) {
            int idx = tid + i * 256;
            int rr = idx >> 2, c8 = idx & 3;
            unsigned lds = (unsigned)(size_t)(&Ash[buf * ASH + rr * 40 + c8 * 8]);
            unsigned go  = (unsigned)(((unsigned)(m0 + rr) * EMB + k0 + c8 * 8) * 2);
            asm volatile("global_load_async_to_lds_b128 %0, %1, %2 offset:0"
                         :: "v"(lds), "v"(go), "s"(abase) : "memory");
        }
    };
    auto fetch = [&](int k0) {
        if constexpr (!A_F16) {
            const float* Af = (const float*)Ag;
#pragma unroll
            for (int i = 0; i < 4; ++i) {
                int idx = tid + i * 256;
                int r = idx >> 3, c4 = idx & 7;
                pfA4[i] = *(const v4f*)(Af + (size_t)(m0 + r) * EMB + k0 + c4 * 4);
            }
        }
#pragma unroll
        for (int i = 0; i < 4; ++i) {
            int idx = tid + i * 256;
            int kr = idx >> 5, c2 = idx & 31;
            pfW[i] = *(const v2f*)(Wg + (size_t)(k0 + kr) * EMB + n0 + c2 * 2);
        }
    };
    auto commit = [&]() {
        if constexpr (!A_F16) {
#pragma unroll
            for (int i = 0; i < 4; ++i) {
                int idx = tid + i * 256;
                int r = idx >> 3, c4 = idx & 7;
                v4h h;
                h[0] = (_Float16)pfA4[i][0]; h[1] = (_Float16)pfA4[i][1];
                h[2] = (_Float16)pfA4[i][2]; h[3] = (_Float16)pfA4[i][3];
                *(v4h*)(&Ash[r * 40 + c4 * 4]) = h;
            }
        }
#pragma unroll
        for (int i = 0; i < 4; ++i) {
            int idx = tid + i * 256;
            int kr = idx >> 5, c2 = idx & 31;
            Bsh[(c2 * 2    ) * 40 + kr] = (_Float16)pfW[i][0];
            Bsh[(c2 * 2 + 1) * 40 + kr] = (_Float16)pfW[i][1];
        }
    };

    if constexpr (A_F16) issueA(0, 0);
    fetch(0);
    int p = 0;
    for (int k0 = 0; k0 < EMB; k0 += 32) {
        const bool more = (k0 + 32 < EMB);
        __syncthreads();                    // prior reads complete
        commit();
        if (more) {
            if constexpr (A_F16) issueA(k0 + 32, p ^ 1);
            fetch(k0 + 32);
        }
        if constexpr (A_F16) {
            if (more) asm volatile("s_wait_asynccnt 0x2" ::: "memory");
            else      asm volatile("s_wait_asynccnt 0x0" ::: "memory");
        }
        __syncthreads();                    // tile visible

        const int arow = wave * 16 + ln;
        const _Float16* Ab = Ash + (A_F16 ? p * ASH : 0);
        v8h alo = *(const v8h*)(&Ab[arow * 40 + 8 * lg]);
        v8h ahi = *(const v8h*)(&Ab[arow * 40 + 16 + 8 * lg]);
        v16h a = cat8(alo, ahi);
#pragma unroll
        for (int t = 0; t < 4; ++t) {
            const int brow = t * 16 + ln;
            v8h blo = *(const v8h*)(&Bsh[brow * 40 + 8 * lg]);
            v8h bhi = *(const v8h*)(&Bsh[brow * 40 + 16 + 8 * lg]);
            acc[t] = wmma_f16(a, cat8(blo, bhi), acc[t]);
        }
        p ^= 1;
    }

    // ---- epilogue: bias+scale, stage in LDS, packed b128 stores ----
    __syncthreads();
    if constexpr (OUT_F32) {
        float* Cw = (float*)pool + wave * (16 * 64);
#pragma unroll
        for (int t = 0; t < 4; ++t) {
            const float bv = bias[n0 + t * 16 + ln];
#pragma unroll
            for (int v = 0; v < 8; ++v)
                Cw[(v + 8 * lg) * 64 + t * 16 + ln] = (acc[t][v] + bv) * scale;
        }
        const int row = m0 + wave * 16 + ln;
        float* og = (float*)Og + (size_t)row * EMB + n0 + lg * 32;
#pragma unroll
        for (int i = 0; i < 8; ++i)
            *(v4f*)(og + i * 4) = *(const v4f*)(&Cw[ln * 64 + lg * 32 + i * 4]);
    } else {
        _Float16* Cw = (_Float16*)pool + wave * (16 * 64);
#pragma unroll
        for (int t = 0; t < 4; ++t) {
            const float bv = bias[n0 + t * 16 + ln];
#pragma unroll
            for (int v = 0; v < 8; ++v)
                Cw[(v + 8 * lg) * 64 + t * 16 + ln] =
                    (_Float16)((acc[t][v] + bv) * scale);
        }
        const int row = m0 + wave * 16 + ln;
        _Float16* og = (_Float16*)Og + (size_t)row * EMB + n0 + lg * 32;
#pragma unroll
        for (int i = 0; i < 4; ++i)
            *(v8h*)(og + i * 8) = *(const v8h*)(&Cw[ln * 64 + lg * 32 + i * 8]);
    }
}

// ---------------------------------------------------------------------------
// Flash attention: grid (qb, h, b); 8 waves x 16 q-rows; key blocks of 64.
// K tile: async global->LDS (double-buffered). V tile: register transpose.
// ---------------------------------------------------------------------------
__global__ __launch_bounds__(256) void attn_kernel(
    const _Float16* __restrict__ Q, const _Float16* __restrict__ K,
    const _Float16* __restrict__ V, _Float16* __restrict__ O)
{
    __shared__ _Float16 Ksh[2][64 * 72];   // [key][d], double-buffered
    __shared__ _Float16 Vsh[64 * 72];      // transposed [d][key]
    __shared__ _Float16 Psh[8][16 * 72];   // per-wave prob/output scratch

    const int tid  = threadIdx.x;
    const int wave = tid >> 5;
    const int lane = tid & 31;
    const int lg   = lane >> 4;
    const int ln   = lane & 15;
    const int qb   = blockIdx.x;
    const int h    = blockIdx.y;
    const int b    = blockIdx.z;
    const int colh = h * HDIM;

    // preload Q A-fragments
    const size_t rowQ = (size_t)(b * SEQ + qb * 128 + wave * 16 + ln);
    v16h qa[2];
#pragma unroll
    for (int kt = 0; kt < 2; ++kt) {
        v8h lo = *(const v8h*)(Q + rowQ * EMB + colh + kt * 32 + 8 * lg);
        v8h hi = *(const v8h*)(Q + rowQ * EMB + colh + kt * 32 + 16 + 8 * lg);
        qa[kt] = cat8(lo, hi);
    }

    v8f o[4] = {};
    float m[8], l[8];
#pragma unroll
    for (int v = 0; v < 8; ++v) { m[v] = -1e30f; l[v] = 0.f; }

    const int r  = tid >> 2;               // staging row (0..63)
    const int ch = tid & 3;                // 16-half chunk
    v8h vreg[2];
    const _Float16* kbase = K + (size_t)b * SEQ * EMB;   // uniform -> SGPR pair

    auto issueK = [&](int j, int buf) {    // async K tile -> LDS (pure copy)
        unsigned lds = (unsigned)(size_t)(&Ksh[buf][r * 72 + ch * 16]);
        unsigned go  = (unsigned)(((unsigned)(j + r) * EMB + colh + ch * 16) * 2);
        // instruction offset applies to BOTH lds and global address
        asm volatile("global_load_async_to_lds_b128 %0, %1, %2 offset:0"
                     :: "v"(lds), "v"(go), "s"(kbase) : "memory");
        asm volatile("global_load_async_to_lds_b128 %0, %1, %2 offset:16"
                     :: "v"(lds), "v"(go), "s"(kbase) : "memory");
    };
    auto fetchV = [&](int j) {
        const _Float16* vg = V + (size_t)(b * SEQ + j + r) * EMB + colh + ch * 16;
        vreg[0] = *(const v8h*)(vg);
        vreg[1] = *(const v8h*)(vg + 8);
    };
    auto commitV = [&]() {
#pragma unroll
        for (int e = 0; e < 8; ++e) Vsh[(ch * 16 + e) * 72 + r]     = vreg[0][e];
#pragma unroll
        for (int e = 0; e < 8; ++e) Vsh[(ch * 16 + 8 + e) * 72 + r] = vreg[1][e];
    };

    issueK(0, 0);
    fetchV(0);
    int p = 0;
    for (int j = 0; j < SEQ; j += 64) {
        const bool more = (j + 64 < SEQ);
        __syncthreads();                   // prior compute done with LDS
        commitV();
        if (more) { issueK(j + 64, p ^ 1); fetchV(j + 64); }
        if (j + 128 < SEQ)                 // L2 prefetch hint for V
            __builtin_prefetch(V + (size_t)(b * SEQ + j + 128 + r) * EMB + colh, 0, 3);
        if (more) asm volatile("s_wait_asynccnt 0x2" ::: "memory");
        else      asm volatile("s_wait_asynccnt 0x0" ::: "memory");
        __syncthreads();                   // K (async) + V tiles visible

        // ---- scores S = Q K^T ----
        const _Float16* Kb = &Ksh[p][0];
        v8f s[4];
#pragma unroll
        for (int t = 0; t < 4; ++t) {
            v8f a0 = {};
#pragma unroll
            for (int kt = 0; kt < 2; ++kt) {
                const int krow = t * 16 + ln;
                v8h blo = *(const v8h*)(&Kb[krow * 72 + kt * 32 + 8 * lg]);
                v8h bhi = *(const v8h*)(&Kb[krow * 72 + kt * 32 + 16 + 8 * lg]);
                a0 = wmma_f16(qa[kt], cat8(blo, bhi), a0);
            }
            s[t] = a0;
        }

        // ---- online softmax (rows = v + 8*lg, cols across 16 lanes) ----
        float newm[8], f[8];
#pragma unroll
        for (int v = 0; v < 8; ++v) {
            float mx = fmaxf(fmaxf(s[0][v], s[1][v]), fmaxf(s[2][v], s[3][v]));
            mx = row16_max(mx);
            newm[v] = fmaxf(m[v], mx);
            f[v] = __expf(m[v] - newm[v]);
            m[v] = newm[v];
        }
        float rs[8];
#pragma unroll
        for (int v = 0; v < 8; ++v) rs[v] = 0.f;
#pragma unroll
        for (int t = 0; t < 4; ++t)
#pragma unroll
            for (int v = 0; v < 8; ++v) {
                float pe = __expf(s[t][v] - newm[v]);
                s[t][v] = pe;
                rs[v] += pe;
            }
#pragma unroll
        for (int v = 0; v < 8; ++v)
            l[v] = l[v] * f[v] + row16_sum(rs[v]);
#pragma unroll
        for (int t = 0; t < 4; ++t)
#pragma unroll
            for (int v = 0; v < 8; ++v) o[t][v] *= f[v];

        // ---- P: C-layout -> LDS -> A-layout (wave-private) ----
        _Float16* Pw = &Psh[wave][0];
#pragma unroll
        for (int t = 0; t < 4; ++t)
#pragma unroll
            for (int v = 0; v < 8; ++v)
                Pw[(v + 8 * lg) * 72 + t * 16 + ln] = (_Float16)s[t][v];

        v16h pa[2];
#pragma unroll
        for (int kt = 0; kt < 2; ++kt) {
            v8h lo = *(const v8h*)(&Pw[ln * 72 + kt * 32 + 8 * lg]);
            v8h hi = *(const v8h*)(&Pw[ln * 72 + kt * 32 + 16 + 8 * lg]);
            pa[kt] = cat8(lo, hi);
        }

        // ---- O += P @ V ----
#pragma unroll
        for (int t = 0; t < 4; ++t)
#pragma unroll
            for (int kt = 0; kt < 2; ++kt) {
                const int vrow = t * 16 + ln;
                v8h blo = *(const v8h*)(&Vsh[vrow * 72 + kt * 32 + 8 * lg]);
                v8h bhi = *(const v8h*)(&Vsh[vrow * 72 + kt * 32 + 16 + 8 * lg]);
                o[t] = wmma_f16(pa[kt], cat8(blo, bhi), o[t]);
            }
        p ^= 1;
    }

    // ---- normalize, stage in wave LDS, packed b128 stores ----
    _Float16* Pw = &Psh[wave][0];
#pragma unroll
    for (int v = 0; v < 8; ++v) {
        const float inv = 1.0f / l[v];
#pragma unroll
        for (int t = 0; t < 4; ++t)
            Pw[(v + 8 * lg) * 72 + t * 16 + ln] = (_Float16)(o[t][v] * inv);
    }
    const size_t rowO = (size_t)(b * SEQ + qb * 128 + wave * 16 + ln);
    _Float16* og = O + rowO * EMB + colh + lg * 32;
#pragma unroll
    for (int i = 0; i < 4; ++i)
        *(v8h*)(og + i * 8) = *(const v8h*)(&Pw[ln * 72 + lg * 32 + i * 8]);
}

// ---------------------------------------------------------------------------
extern "C" void kernel_launch(void* const* d_in, const int* in_sizes, int n_in,
                              void* d_out, int out_size, void* d_ws, size_t ws_size,
                              hipStream_t stream) {
    (void)in_sizes; (void)n_in; (void)out_size; (void)ws_size;
    const float* query    = (const float*)d_in[0];
    const float* key      = (const float*)d_in[1];
    const float* value    = (const float*)d_in[2];
    const float* W_Query  = (const float*)d_in[3];
    const float* W_Key    = (const float*)d_in[4];
    const float* W_Value  = (const float*)d_in[5];
    const float* W_Output = (const float*)d_in[6];
    const float* B_Query  = (const float*)d_in[7];
    const float* B_Key    = (const float*)d_in[8];
    const float* B_Value  = (const float*)d_in[9];
    const float* B_Output = (const float*)d_in[10];

    const size_t mat = (size_t)MROWS * EMB;
    _Float16* Qh = (_Float16*)d_ws;
    _Float16* Kh = Qh + mat;
    _Float16* Vh = Kh + mat;
    _Float16* Oh = Vh + mat;

    dim3 blk(256);
    dim3 gg(MROWS / 128, EMB / 64);

    gemm_kernel<false, false><<<gg, blk, 0, stream>>>(query, W_Query, B_Query, Qh, 0.125f);
    gemm_kernel<false, false><<<gg, blk, 0, stream>>>(key,   W_Key,   B_Key,   Kh, 1.0f);
    gemm_kernel<false, false><<<gg, blk, 0, stream>>>(value, W_Value, B_Value, Vh, 1.0f);

    attn_kernel<<<dim3(SEQ / 128, HEADS, NB), blk, 0, stream>>>(Qh, Kh, Vh, Oh);

    gemm_kernel<true, true><<<gg, blk, 0, stream>>>(Oh, W_Output, B_Output, (float*)d_out, 1.0f);
}
